// TFAttention_68788196213308
// MI455X (gfx1250) — compile-verified, matrix operand order checked by reading, not run on hardware
//
#include <hip/hip_runtime.h>
#include <hip/hip_bf16.h>
#include <math.h>

// Problem constants (match reference)
#define BB  4
#define CC  128
#define TT  1000
#define FF  129
#define HH  4
#define HCC 4
#define VCC 32
#define HB  (HH * BB)        // 16 (h,b) pairs
#define TP  1024             // T padded to mult of 32 (K dim of P@V)
#define DQ  544              // HC*F = 516 padded to mult of 32 (K dim of Q@K^T)
#define DQV 516
#define DV  4128             // VC*F (mult of 16, N dim of P@V)
#define EPSN 1e-5f

typedef __attribute__((ext_vector_type(16))) __bf16       v16bf;
typedef __attribute__((ext_vector_type(8)))  float        v8f;
typedef __attribute__((ext_vector_type(4)))  unsigned int uint4v;
typedef __hip_bfloat16 bf16;

union FragBF { v16bf bf; uint4v u[2]; };

// Load a 16-element bf16 WMMA A/B fragment for this lane.
// rowPtr = start of this lane's (row-major, K-contiguous) matrix row.
// k0     = k-step base (mult of 32) + per-lane sub-offset ((lane>>4)*8).
// Per the CDNA5 16-bit A/B layout, lane holds K chunks [k0..k0+7] and
// [k0+16..k0+23]; two 16-byte loads.
static __device__ inline v16bf load_frag(const bf16* __restrict__ rowPtr, int k0) {
    FragBF f;
    f.u[0] = *reinterpret_cast<const uint4v*>(rowPtr + k0);
    f.u[1] = *reinterpret_cast<const uint4v*>(rowPtr + k0 + 16);
    return f.bf;
}

// -------------------------------------------------------------------------
// Kernel 1: fused QKV projection + PReLU + per-head ChannelNorm.
// Grid = BB*TP blocks of 256. Blocks with t>=TT zero-fill all pad rows/cols.
// Outputs: qf/kf row-major [hb][TP][DQ] bf16, v transposed vfT [hb][DV][TP].
// -------------------------------------------------------------------------
static __device__ inline float k1_dot(const float* __restrict__ wA,
                                      const float* __restrict__ xs,
                                      int o, int f,
                                      const float* __restrict__ bq,
                                      const float* __restrict__ bk,
                                      const float* __restrict__ bv,
                                      const float* __restrict__ aq,
                                      const float* __restrict__ ak,
                                      const float* __restrict__ av,
                                      int& g)
{
    float acc, slope;
    if (o < 16)      { acc = bq[o];      slope = aq[o >> 2];        g = o >> 2; }
    else if (o < 32) { acc = bk[o - 16]; slope = ak[(o - 16) >> 2]; g = 4 + ((o - 16) >> 2); }
    else             { acc = bv[o - 32]; slope = av[(o - 32) >> 5]; g = 8 + ((o - 32) >> 5); }
    const float* wr = wA + o * CC;
#pragma unroll 8
    for (int c = 0; c < CC; ++c) acc = fmaf(wr[c], xs[c * 132 + f], acc);
    return (acc >= 0.f) ? acc : slope * acc;
}

__global__ __launch_bounds__(256)
void k1_qkv(const float* __restrict__ x,
            const float* __restrict__ Wq, const float* __restrict__ bq,
            const float* __restrict__ aq, const float* __restrict__ gq,
            const float* __restrict__ btq,
            const float* __restrict__ Wk, const float* __restrict__ bk,
            const float* __restrict__ ak, const float* __restrict__ gk,
            const float* __restrict__ btk,
            const float* __restrict__ Wv, const float* __restrict__ bv,
            const float* __restrict__ av, const float* __restrict__ gv,
            const float* __restrict__ btv,
            bf16* __restrict__ qf, bf16* __restrict__ kf,
            bf16* __restrict__ vfT)
{
    const int tid = threadIdx.x;
    const int b   = blockIdx.x >> 10;       // TP == 1024
    const int t   = blockIdx.x & (TP - 1);
    const bf16 zero = __float2bfloat16(0.0f);

    if (t >= TT) {
        // zero padded time rows of qf/kf and padded time columns of vfT
        for (int h = 0; h < HH; ++h) {
            size_t row = ((size_t)(h * BB + b) * TP + t) * DQ;
            for (int i = tid; i < DQ; i += 256) { qf[row + i] = zero; kf[row + i] = zero; }
        }
        for (int i = tid; i < HH * DV; i += 256) {
            int h = i / DV, e = i - h * DV;
            vfT[((size_t)(h * BB + b) * DV + e) * TP + t] = zero;
        }
        return;
    }

    __shared__ float xs[CC * 132];      // x[:, t, :] column, padded stride
    __shared__ float wA[160 * CC];      // [Wq;Wk;Wv] stacked: 160 x 128
    __shared__ float stats[24];         // 12 groups x {sum, sumsq}

    for (int i = tid; i < CC * FF; i += 256) {
        int c = i / FF, f = i - c * FF;
        xs[c * 132 + f] = x[(((size_t)b * CC + c) * TT + t) * FF + f];
    }
    for (int i = tid; i < 16 * CC; i += 256) {
        wA[i] = Wq[i];
        wA[16 * CC + i] = Wk[i];
    }
    for (int i = tid; i < CC * CC; i += 256) wA[32 * CC + i] = Wv[i];
    if (tid < 24) stats[tid] = 0.f;
    __syncthreads();

    const int NV = 160 * FF;            // 20640 outputs per (b,t)

    // Pass 1: accumulate per-group sum / sumsq (flush on group change).
    {
        int curg = -1; float s = 0.f, sq = 0.f;
        for (int idx = tid; idx < NV; idx += 256) {
            int o = idx / FF, f = idx - o * FF;
            int g;
            float acc = k1_dot(wA, xs, o, f, bq, bk, bv, aq, ak, av, g);
            if (g != curg) {
                if (curg >= 0) {
                    atomicAdd(&stats[2 * curg], s);
                    atomicAdd(&stats[2 * curg + 1], sq);
                }
                curg = g; s = 0.f; sq = 0.f;
            }
            s += acc; sq += acc * acc;
        }
        if (curg >= 0) {
            atomicAdd(&stats[2 * curg], s);
            atomicAdd(&stats[2 * curg + 1], sq);
        }
    }
    __syncthreads();

    // Pass 2: recompute, normalize, affine, store as bf16.
    for (int idx = tid; idx < NV; idx += 256) {
        int o = idx / FF, f = idx - o * FF;
        int g;
        float acc = k1_dot(wA, xs, o, f, bq, bk, bv, aq, ak, av, g);
        float n   = (o < 32) ? (float)(HCC * FF) : (float)(VCC * FF);
        float mu  = stats[2 * g] / n;
        float var = stats[2 * g + 1] / n - mu * mu;
        float rs  = rsqrtf(var + EPSN);
        float ga, bt2;
        if (o < 16)      { ga = gq[o * FF + f];        bt2 = btq[o * FF + f]; }
        else if (o < 32) { ga = gk[(o - 16) * FF + f]; bt2 = btk[(o - 16) * FF + f]; }
        else             { ga = gv[(o - 32) * FF + f]; bt2 = btv[(o - 32) * FF + f]; }
        float val = (acc - mu) * rs * ga + bt2;
        bf16 vb = __float2bfloat16(val);
        if (o < 16) {
            int h = o >> 2, hc = o & 3;
            qf[((size_t)(h * BB + b) * TP + t) * DQ + hc * FF + f] = vb;
        } else if (o < 32) {
            int oo = o - 16, h = oo >> 2, hc = oo & 3;
            kf[((size_t)(h * BB + b) * TP + t) * DQ + hc * FF + f] = vb;
        } else {
            int oo = o - 32, h = oo >> 5, vc = oo & 31;
            vfT[((size_t)(h * BB + b) * DV + vc * FF + f) * TP + t] = vb;
        }
    }

    // Zero qf/kf K-pad columns [516,544) for this t row.
    for (int i = tid; i < 8 * (DQ - DQV); i += 256) {
        int which = i / (DQ - DQV);
        int dd    = DQV + i % (DQ - DQV);
        int h     = which & 3;
        size_t row = ((size_t)(h * BB + b) * TP + t) * DQ + dd;
        if (which >> 2) kf[row] = zero; else qf[row] = zero;
    }
}

// -------------------------------------------------------------------------
// Kernel 2a: S = scale * Q @ K^T (bf16 WMMA, fp32 acc), row softmax -> P bf16.
// Grid = (63 row tiles, 16 hb). One block = 16 query rows, 8 waves cover the
// 63 column tiles; S staged in LDS for the softmax.
// -------------------------------------------------------------------------
#define SL 1008   // 63 * 16 columns

__global__ __launch_bounds__(256)
void k2a_scores(const bf16* __restrict__ qf, const bf16* __restrict__ kf,
                bf16* __restrict__ P)
{
    __shared__ float S[16 * SL];
    __shared__ float red[256];

    const int tid  = threadIdx.x;
    const int lane = tid & 31;
    const int wave = tid >> 5;
    const int hb   = blockIdx.y;
    const int rowBase = blockIdx.x * 16;
    const float scale = rsqrtf((float)(HCC * FF));   // 1/sqrt(516)
    const int subk = (lane >> 4) << 3;

    const bf16* arow = qf + ((size_t)hb * TP + rowBase + (lane & 15)) * DQ;

    for (int ct = wave; ct < 63; ct += 8) {
        const bf16* brow = kf + ((size_t)hb * TP + ct * 16 + (lane & 15)) * DQ;
        v8f c = {0.f, 0.f, 0.f, 0.f, 0.f, 0.f, 0.f, 0.f};
        for (int ks = 0; ks < DQ / 32; ++ks) {
            int k0 = ks * 32 + subk;
            v16bf a  = load_frag(arow, k0);
            v16bf bm = load_frag(brow, k0);
            c = __builtin_amdgcn_wmma_f32_16x16x32_bf16(
                    false, a, false, bm, (short)0, c, false, false);
        }
        int col  = ct * 16 + (lane & 15);
        int rofs = (lane >> 4) * 8;
#pragma unroll
        for (int i = 0; i < 8; ++i) S[(rofs + i) * SL + col] = c[i] * scale;
    }
    __syncthreads();

    // Row softmax over s < TT (mask pads), 16 threads per row.
    const int r  = tid >> 4;
    const int wi = tid & 15;

    float m = -INFINITY;
    for (int s = wi; s < TT; s += 16) m = fmaxf(m, S[r * SL + s]);
    red[tid] = m;
    __syncthreads();
    for (int st = 8; st > 0; st >>= 1) {
        if (wi < st) red[tid] = fmaxf(red[tid], red[tid + st]);
        __syncthreads();
    }
    float rowm = red[r << 4];
    __syncthreads();

    float sum = 0.f;
    for (int s = wi; s < TT; s += 16) {
        float e = __expf(S[r * SL + s] - rowm);
        S[r * SL + s] = e;
        sum += e;
    }
    red[tid] = sum;
    __syncthreads();
    for (int st = 8; st > 0; st >>= 1) {
        if (wi < st) red[tid] += red[tid + st];
        __syncthreads();
    }
    float inv = 1.f / red[r << 4];

    int trow = rowBase + r;
    if (trow < TT) {
        bf16* prow = P + ((size_t)hb * TP + trow) * TP;
        const bf16 zero = __float2bfloat16(0.0f);
        for (int s = wi; s < TP; s += 16)
            prow[s] = (s < TT) ? __float2bfloat16(S[r * SL + s] * inv) : zero;
    }
}

// -------------------------------------------------------------------------
// Kernel 2b: out2 = P @ V  (bf16 WMMA, fp32 acc). Each wave owns one 16x16
// output tile; B operand reads pre-transposed vfT so K is contiguous.
// Grid = (33 col-tile groups, 63 row tiles, 16 hb), 8 waves/block.
// -------------------------------------------------------------------------
__global__ __launch_bounds__(256)
void k2b_pv(const bf16* __restrict__ P, const bf16* __restrict__ vfT,
            float* __restrict__ out2)
{
    const int tid  = threadIdx.x;
    const int lane = tid & 31;
    const int wave = tid >> 5;
    const int hb   = blockIdx.z;
    const int rowBase = blockIdx.y * 16;
    const int ct   = blockIdx.x * 8 + wave;
    if (ct >= DV / 16) return;
    const int subk = (lane >> 4) << 3;

    const bf16* arow = P   + ((size_t)hb * TP + rowBase + (lane & 15)) * TP;
    const bf16* brow = vfT + ((size_t)hb * DV + ct * 16 + (lane & 15)) * TP;

    v8f c = {0.f, 0.f, 0.f, 0.f, 0.f, 0.f, 0.f, 0.f};
    for (int ks = 0; ks < TP / 32; ++ks) {
        int k0 = ks * 32 + subk;
        v16bf a  = load_frag(arow, k0);
        v16bf bm = load_frag(brow, k0);
        c = __builtin_amdgcn_wmma_f32_16x16x32_bf16(
                false, a, false, bm, (short)0, c, false, false);
    }

    int col  = ct * 16 + (lane & 15);
    int rofs = (lane >> 4) * 8;
#pragma unroll
    for (int i = 0; i < 8; ++i) {
        int row = rowBase + rofs + i;
        if (row < TT) out2[((size_t)hb * TT + row) * DV + col] = c[i];
    }
}

// -------------------------------------------------------------------------
// Kernel 3: output projection + PReLU + ChannelNorm + residual.
// Grid = BB*TT blocks of 256, same LDS-staged scheme as kernel 1.
// -------------------------------------------------------------------------
__global__ __launch_bounds__(256)
void k3_proj(const float* __restrict__ out2, const float* __restrict__ x,
             const float* __restrict__ Wp, const float* __restrict__ bp,
             const float* __restrict__ ap, const float* __restrict__ gp,
             const float* __restrict__ btp, float* __restrict__ out)
{
    __shared__ float as[CC * 132];
    __shared__ float wp[CC * CC];
    __shared__ float st[2];

    const int tid = threadIdx.x;
    const int b   = blockIdx.x / TT;
    const int t   = blockIdx.x - b * TT;

    // gather attention output column: channel c = h*VC + vc
    for (int i = tid; i < CC * FF; i += 256) {
        int c = i / FF, f = i - c * FF;
        int h = c >> 5, vc = c & 31;
        as[c * 132 + f] = out2[((size_t)(h * BB + b) * TT + t) * DV + vc * FF + f];
    }
    for (int i = tid; i < CC * CC; i += 256) wp[i] = Wp[i];
    if (tid < 2) st[tid] = 0.f;
    __syncthreads();

    const float slope = ap[0];
    const int NV = CC * FF;   // 16512

    float s = 0.f, sq = 0.f;
    for (int idx = tid; idx < NV; idx += 256) {
        int o = idx / FF, f = idx - o * FF;
        float acc = bp[o];
        const float* wr = &wp[o * CC];
#pragma unroll 8
        for (int c = 0; c < CC; ++c) acc = fmaf(wr[c], as[c * 132 + f], acc);
        acc = (acc >= 0.f) ? acc : slope * acc;
        s += acc; sq += acc * acc;
    }
    atomicAdd(&st[0], s);
    atomicAdd(&st[1], sq);
    __syncthreads();

    float n   = (float)NV;
    float mu  = st[0] / n;
    float var = st[1] / n - mu * mu;
    float rs  = rsqrtf(var + EPSN);

    for (int idx = tid; idx < NV; idx += 256) {
        int o = idx / FF, f = idx - o * FF;
        float acc = bp[o];
        const float* wr = &wp[o * CC];
#pragma unroll 8
        for (int c = 0; c < CC; ++c) acc = fmaf(wr[c], as[c * 132 + f], acc);
        acc = (acc >= 0.f) ? acc : slope * acc;
        size_t gi = (((size_t)b * CC + o) * TT + t) * FF + f;
        out[gi] = (acc - mu) * rs * gp[idx] + btp[idx] + x[gi];
    }
}

// -------------------------------------------------------------------------
// Host entry.
// Inputs (setup_inputs order): x, Wq,bq,aq,gq,btq, Wk,bk,ak,gk,btk,
// Wv,bv,av,gv,btv, Wp,bp,ap,gp,btp. All fp32.
// Workspace (~470 MB): qf, kf, vfT, P, out2.
// -------------------------------------------------------------------------
extern "C" void kernel_launch(void* const* d_in, const int* in_sizes, int n_in,
                              void* d_out, int out_size, void* d_ws, size_t ws_size,
                              hipStream_t stream)
{
    const float* x   = (const float*)d_in[0];
    const float* Wq  = (const float*)d_in[1];
    const float* bq  = (const float*)d_in[2];
    const float* aq  = (const float*)d_in[3];
    const float* gq  = (const float*)d_in[4];
    const float* btq = (const float*)d_in[5];
    const float* Wk  = (const float*)d_in[6];
    const float* bk  = (const float*)d_in[7];
    const float* ak  = (const float*)d_in[8];
    const float* gk  = (const float*)d_in[9];
    const float* btk = (const float*)d_in[10];
    const float* Wv  = (const float*)d_in[11];
    const float* bv  = (const float*)d_in[12];
    const float* av  = (const float*)d_in[13];
    const float* gv  = (const float*)d_in[14];
    const float* btv = (const float*)d_in[15];
    const float* Wp  = (const float*)d_in[16];
    const float* bp  = (const float*)d_in[17];
    const float* ap  = (const float*)d_in[18];
    const float* gp  = (const float*)d_in[19];
    const float* btp = (const float*)d_in[20];

    char* ws = (char*)d_ws;
    size_t off = 0;
    auto take = [&](size_t bytes) -> char* {
        char* p = ws + off;
        off += (bytes + 255) & ~(size_t)255;
        return p;
    };
    bf16*  qf   = (bf16*)take((size_t)HB * TP * DQ * sizeof(bf16));
    bf16*  kf   = (bf16*)take((size_t)HB * TP * DQ * sizeof(bf16));
    bf16*  vfT  = (bf16*)take((size_t)HB * DV * TP * sizeof(bf16));
    bf16*  Pm   = (bf16*)take((size_t)HB * TP * TP * sizeof(bf16));
    float* o2   = (float*)take((size_t)HB * TT * DV * sizeof(float));

    k1_qkv<<<BB * TP, 256, 0, stream>>>(x, Wq, bq, aq, gq, btq,
                                        Wk, bk, ak, gk, btk,
                                        Wv, bv, av, gv, btv,
                                        qf, kf, vfT);
    k2a_scores<<<dim3(63, HB), 256, 0, stream>>>(qf, kf, Pm);
    k2b_pv<<<dim3((DV / 16 + 7) / 8, 63, HB), 256, 0, stream>>>(Pm, vfT, o2);
    k3_proj<<<BB * TT, 256, 0, stream>>>(o2, x, Wp, bp, ap, gp, btp,
                                         (float*)d_out);
}
